// ShowAttendTellDecoder_8735963480727
// MI455X (gfx1250) — matrix operand order repaired and checked by reading, not compiled, not run on hardware
//
#include <hip/hip_runtime.h>
#include <hip/hip_bf16.h>
#include <math.h>

typedef __bf16 bf16_t;
typedef __attribute__((ext_vector_type(16))) __bf16        v16bf;
typedef __attribute__((ext_vector_type(8)))  float         v8f;
typedef __attribute__((ext_vector_type(4)))  unsigned int  u32x4;

#define B_   64
#define C_   1024
#define HW_  256
#define T_   100
#define V_   10000
#define E_   512
#define HD_  512
#define AP_  112   /* attention dim A=100 padded to mult of 16 */

union Frag16 { v16bf v; u32x4 q[2]; };

__device__ __forceinline__ float sigm(float x){ return 1.0f/(1.0f+expf(-x)); }

// ---------------------------------------------------------------------------
// Generic bf16 WMMA GEMM: D(MxN) = A(MxK) * Bpacked(KxN) + epilogue
// One wave per 16x16 tile; 4 waves (128 thr) per block along N.
// mode 0: outf[m*ldo+n] = acc + bias[n]
// mode 1: outf[m*ldo+n] = tanh(acc + bias[n])
// mode 2: outb[m*ldo+n] = (bf16) tanh(acc + bias[n] + extra[m*lde+n])
// ---------------------------------------------------------------------------
__global__ __launch_bounds__(128)
void wmma_gemm(const bf16_t* __restrict__ A, int lda,
               const bf16_t* __restrict__ Bpk, int nTilesN,
               const float* __restrict__ bias,
               const float* __restrict__ extra, int lde,
               float* __restrict__ outf, bf16_t* __restrict__ outb,
               long long ldo, int K, int mode)
{
    const int wave = threadIdx.x >> 5;
    const int lane = threadIdx.x & 31;
    const int mt   = blockIdx.x;
    const int nt   = blockIdx.y * 4 + wave;
    if (nt >= nTilesN) return;                 // wave-uniform exit

    const int half = lane >> 4;                // 0: K 0..15 ; 1: K 16..31
    const int l16  = lane & 15;

    const bf16_t* arow = A + (size_t)(mt * 16 + l16) * lda + half * 16;
    const bf16_t* bptr = Bpk + (size_t)nt * 512 + lane * 16;
    const size_t  bstr = (size_t)nTilesN * 512;

    v8f acc = {};
    for (int k0 = 0; k0 < K; k0 += 32) {
        Frag16 a, b;
        a.q[0] = *(const u32x4*)(arow + k0);
        a.q[1] = *(const u32x4*)(arow + k0 + 8);
        b.q[0] = *(const u32x4*)(bptr);
        b.q[1] = *(const u32x4*)(bptr + 8);
        bptr += bstr;
        acc = __builtin_amdgcn_wmma_f32_16x16x32_bf16(
                  false, a.v, false, b.v, (short)0, acc, false, false);
    }

    const int n     = nt * 16 + l16;
    const int mbase = mt * 16 + half * 8;
    const float bn  = bias ? bias[n] : 0.0f;
#pragma unroll
    for (int i = 0; i < 8; ++i) {
        const int m = mbase + i;
        float v = acc[i] + bn;
        if (mode == 0) {
            outf[(size_t)m * ldo + n] = v;
        } else if (mode == 1) {
            outf[(size_t)m * ldo + n] = tanhf(v);
        } else {
            v = tanhf(v + extra[(size_t)m * lde + n]);
            outb[(size_t)m * ldo + n] = (bf16_t)v;
        }
    }
}

// ---------------------------------------------------------------------------
// Pack fp32 weights (K x N row-major, optionally vertically stacked W1/W2)
// into WMMA B-fragment order: per (kt,nt) tile, lane L<16 holds col nt*16+L,
// K rows kt*32+0..15 ; lane>=16 holds same cols, K rows kt*32+16..31.
// ---------------------------------------------------------------------------
__global__ void pack_b(const float* __restrict__ W1, int K1,
                       const float* __restrict__ W2,
                       bf16_t* __restrict__ out, int N, int Npad, size_t total)
{
    size_t p = (size_t)blockIdx.x * 256 + threadIdx.x;
    if (p >= total) return;
    const int within = (int)(p & 511);
    const size_t tile = p >> 9;
    const int lane = within >> 4, elem = within & 15;
    const int ntiles = Npad / 16;
    const int kt = (int)(tile / ntiles), nt = (int)(tile % ntiles);
    const int k = kt * 32 + (lane >> 4) * 16 + elem;
    const int n = nt * 16 + (lane & 15);
    float v = 0.0f;
    if (n < N) v = (k < K1) ? W1[(size_t)k * N + n] : W2[(size_t)(k - K1) * N + n];
    out[p] = (bf16_t)v;
}

// (B,C,H,W) -> feats_t (B,HW,C) fp32 + bf16
__global__ void prep_feats(const float* __restrict__ img,
                           float* __restrict__ ft, bf16_t* __restrict__ fb)
{
    size_t i = (size_t)blockIdx.x * 256 + threadIdx.x;
    if (i >= (size_t)B_ * C_ * HW_) return;
    const int l = (int)(i & (HW_ - 1));
    const size_t r = i >> 8;
    const int c = (int)(r & (C_ - 1));
    const int b = (int)(r >> 10);
    const float v = img[i];
    const size_t o = ((size_t)b * HW_ + l) * C_ + c;
    ft[o] = v; fb[o] = (bf16_t)v;
}

__global__ void feat_mean(const float* __restrict__ ft, bf16_t* __restrict__ fm)
{
    const int b = blockIdx.x, tid = threadIdx.x;   // 256 threads
    for (int c = tid; c < C_; c += 256) {
        float s = 0.0f;
        const float* f = ft + (size_t)b * HW_ * C_ + c;
        for (int l = 0; l < HW_; ++l) s += f[(size_t)l * C_];
        fm[b * C_ + c] = (bf16_t)(s * (1.0f / HW_));
    }
}

// tokens = [START, reports[:,:-1]], embed + batchnorm -> xf/xb (T,B,E)
__global__ void prep_emb(const int* __restrict__ reports,
                         const float* __restrict__ table,
                         const float* __restrict__ g, const float* __restrict__ be,
                         const float* __restrict__ mu, const float* __restrict__ var,
                         float* __restrict__ xf, bf16_t* __restrict__ xb)
{
    size_t i = (size_t)blockIdx.x * 256 + threadIdx.x;
    if (i >= (size_t)T_ * B_ * E_) return;
    const int e = (int)(i & (E_ - 1));
    const size_t r = i >> 9;
    const int b = (int)(r & (B_ - 1));
    const int t = (int)(r >> 6);
    const int tok = (t == 0) ? 1 : reports[b * T_ + (t - 1)];
    const float v = (table[(size_t)tok * E_ + e] - mu[e]) * rsqrtf(var[e] + 1e-5f) * g[e] + be[e];
    xf[i] = v; xb[i] = (bf16_t)v;
}

__global__ void bias_prep(const float* bih, const float* bhh,
                          const float* blz, const float* blh, const float* baf,
                          float* bg, float* bL, float* bafp)
{
    const int i = blockIdx.x * 256 + threadIdx.x;
    if (i < 4 * HD_) bg[i] = bih[i] + bhh[i];
    if (i < E_)      bL[i] = blz[i] + blh[i];
    if (i < AP_)     bafp[i] = (i < 100) ? baf[i] : 0.0f;
}

__global__ void init_split(const float* __restrict__ init_f,
                           float* __restrict__ hf, float* __restrict__ cf,
                           bf16_t* __restrict__ abuf)
{
    const int b = blockIdx.x, e = threadIdx.x;   // 512
    const float h = init_f[b * 2 * HD_ + e];
    const float c = init_f[b * 2 * HD_ + HD_ + e];
    hf[b * HD_ + e] = h; cf[b * HD_ + e] = c;
    abuf[b * 2048 + 1536 + e] = (bf16_t)h;
}

// Attention + softmax + context + fbeta gate; fills abuf [x | gated] slots.
__global__ __launch_bounds__(512)
void step_attn(const float* __restrict__ hf,
               const float* __restrict__ Wah, const float* __restrict__ bah,
               const float* __restrict__ Wav, const float* __restrict__ bav,
               const float* __restrict__ Wfb, const float* __restrict__ bfb,
               const float* __restrict__ feat_att, const float* __restrict__ ft,
               const bf16_t* __restrict__ xb_t, bf16_t* __restrict__ abuf,
               float* __restrict__ scores, int t)
{
    const int b = blockIdx.x, tid = threadIdx.x;
    __shared__ float atth[AP_];
    __shared__ float alpha[HW_];
    __shared__ float red[HW_];
    __shared__ float fgate_s, mx_s, sum_s;
    const float* hb = hf + b * HD_;

    if (tid < 100) {
        float s = bah[tid];
        for (int k = 0; k < HD_; ++k) s += hb[k] * Wah[k * 100 + tid];
        atth[tid] = s;
    } else if (tid < AP_) atth[tid] = 0.0f;
    if (tid == 128) {
        float s = bfb[0];
        for (int k = 0; k < HD_; ++k) s += hb[k] * Wfb[k];
        fgate_s = sigm(s);
    }
    // copy x_t (bf16) into A-buffer
    abuf[b * 2048 + tid] = xb_t[b * E_ + tid];
    __syncthreads();

    if (tid < HW_) {
        const float* fa = feat_att + ((size_t)b * HW_ + tid) * AP_;
        float s = bav[0];
        for (int a = 0; a < 100; ++a) s += tanhf(fa[a] + atth[a]) * Wav[a];
        alpha[tid] = s; red[tid] = s;
    }
    __syncthreads();
    for (int off = 128; off > 0; off >>= 1) {
        if (tid < off) red[tid] = fmaxf(red[tid], red[tid + off]);
        __syncthreads();
    }
    if (tid == 0) mx_s = red[0];
    __syncthreads();
    if (tid < HW_) { float ex = expf(alpha[tid] - mx_s); alpha[tid] = ex; red[tid] = ex; }
    __syncthreads();
    for (int off = 128; off > 0; off >>= 1) {
        if (tid < off) red[tid] += red[tid + off];
        __syncthreads();
    }
    if (tid == 0) sum_s = red[0];
    __syncthreads();
    if (tid < HW_) {
        const float al = alpha[tid] / sum_s;
        alpha[tid] = al;
        scores[(size_t)b * T_ * HW_ + (size_t)t * HW_ + tid] = al;
    }
    __syncthreads();
    // ctx & gated -> abuf[512..1535]
    for (int c = tid; c < C_; c += 512) {
        float s = 0.0f;
        const float* f = ft + (size_t)b * HW_ * C_ + c;
        for (int l = 0; l < HW_; ++l) s += alpha[l] * f[(size_t)l * C_];
        abuf[b * 2048 + 512 + c] = (bf16_t)(fgate_s * s);
    }
}

__global__ void lstm_pw(const float* __restrict__ gates,
                        float* __restrict__ hf, float* __restrict__ cf,
                        bf16_t* __restrict__ abuf)
{
    const int b = blockIdx.x, e = threadIdx.x;   // 512
    const float* g = gates + (size_t)b * 4 * HD_;
    const float ig = sigm(g[e]);
    const float fg = sigm(g[HD_ + e]);
    const float gg = tanhf(g[2 * HD_ + e]);
    const float og = sigm(g[3 * HD_ + e]);
    const float cn = fg * cf[b * HD_ + e] + ig * gg;
    const float hn = og * tanhf(cn);
    cf[b * HD_ + e] = cn;
    hf[b * HD_ + e] = hn;
    abuf[b * 2048 + 1536 + e] = (bf16_t)hn;
}

static inline char* ws_take(char*& p, size_t bytes) {
    char* r = p;
    p += (bytes + 255) & ~(size_t)255;
    return r;
}

extern "C" void kernel_launch(void* const* d_in, const int* in_sizes, int n_in,
                              void* d_out, int out_size, void* d_ws, size_t ws_size,
                              hipStream_t stream)
{
    (void)in_sizes; (void)n_in; (void)out_size; (void)ws_size;
    const float* img    = (const float*)d_in[0];
    const int*   rep    = (const int*)d_in[1];
    const float* Wfc    = (const float*)d_in[2];
    const float* bfc    = (const float*)d_in[3];
    const float* emb    = (const float*)d_in[4];
    const float* bng    = (const float*)d_in[5];
    const float* bnb    = (const float*)d_in[6];
    const float* bnm    = (const float*)d_in[7];
    const float* bnv    = (const float*)d_in[8];
    const float* Waf    = (const float*)d_in[9];
    const float* baf    = (const float*)d_in[10];
    const float* Wah    = (const float*)d_in[11];
    const float* bah    = (const float*)d_in[12];
    const float* Wav    = (const float*)d_in[13];
    const float* bav    = (const float*)d_in[14];
    const float* Wfb    = (const float*)d_in[15];
    const float* bfb    = (const float*)d_in[16];
    const float* Wih    = (const float*)d_in[17];
    const float* bih    = (const float*)d_in[18];
    const float* Whh    = (const float*)d_in[19];
    const float* bhh    = (const float*)d_in[20];
    const float* WLh    = (const float*)d_in[21];
    const float* bLh    = (const float*)d_in[22];
    const float* WLz    = (const float*)d_in[23];
    const float* bLz    = (const float*)d_in[24];
    const float* WLo    = (const float*)d_in[25];
    const float* bLo    = (const float*)d_in[26];

    float* out_seq    = (float*)d_out;                               // (B,T,V)
    float* out_scores = (float*)d_out + (size_t)B_ * T_ * V_;        // (B,T,HW)

    // workspace layout
    char* p = (char*)d_ws;
    float*  ft      = (float*)ws_take(p, (size_t)B_*HW_*C_*4);
    bf16_t* fb      = (bf16_t*)ws_take(p, (size_t)B_*HW_*C_*2);
    float*  xf      = (float*)ws_take(p, (size_t)T_*B_*E_*4);
    bf16_t* xb      = (bf16_t*)ws_take(p, (size_t)T_*B_*E_*2);
    float*  fatt    = (float*)ws_take(p, (size_t)B_*HW_*AP_*4);
    bf16_t* Wg_pk   = (bf16_t*)ws_take(p, (size_t)2048*2048*2);
    bf16_t* WL_pk   = (bf16_t*)ws_take(p, (size_t)1536*512*2);
    bf16_t* WLo_pk  = (bf16_t*)ws_take(p, (size_t)512*10000*2);
    bf16_t* Wfc_pk  = (bf16_t*)ws_take(p, (size_t)1024*1024*2);
    bf16_t* Waf_pk  = (bf16_t*)ws_take(p, (size_t)1024*AP_*2);
    bf16_t* abuf    = (bf16_t*)ws_take(p, (size_t)B_*2048*2);
    float*  gates   = (float*)ws_take(p, (size_t)B_*2048*4);
    float*  hf      = (float*)ws_take(p, (size_t)B_*HD_*4);
    float*  cf      = (float*)ws_take(p, (size_t)B_*HD_*4);
    bf16_t* pre_bf  = (bf16_t*)ws_take(p, (size_t)B_*E_*2);
    float*  init_f  = (float*)ws_take(p, (size_t)B_*2*HD_*4);
    bf16_t* fm_bf   = (bf16_t*)ws_take(p, (size_t)B_*C_*2);
    float*  bias_g  = (float*)ws_take(p, 4*HD_*4);
    float*  bias_L  = (float*)ws_take(p, E_*4);
    float*  bafp    = (float*)ws_take(p, AP_*4);

    // ---------------- setup ----------------
    {
        size_t n = (size_t)B_*C_*HW_;
        prep_feats<<<dim3((unsigned)((n+255)/256)), 256, 0, stream>>>(img, ft, fb);
    }
    feat_mean<<<B_, 256, 0, stream>>>(ft, fm_bf);
    {
        size_t n = (size_t)T_*B_*E_;
        prep_emb<<<dim3((unsigned)((n+255)/256)), 256, 0, stream>>>(rep, emb, bng, bnb, bnm, bnv, xf, xb);
    }
    bias_prep<<<8, 256, 0, stream>>>(bih, bhh, bLz, bLh, baf, bias_g, bias_L, bafp);

    // weight packing (once)
    {
        size_t tot = (size_t)(2048/32)*(2048/16)*512;
        pack_b<<<dim3((unsigned)((tot+255)/256)), 256, 0, stream>>>(Wih, 1536, Whh, Wg_pk, 2048, 2048, tot);
    }
    {
        size_t tot = (size_t)(1536/32)*(512/16)*512;
        pack_b<<<dim3((unsigned)((tot+255)/256)), 256, 0, stream>>>(WLz, 1024, WLh, WL_pk, 512, 512, tot);
    }
    {
        size_t tot = (size_t)(512/32)*(10000/16)*512;
        pack_b<<<dim3((unsigned)((tot+255)/256)), 256, 0, stream>>>(WLo, 512, nullptr, WLo_pk, 10000, 10000, tot);
    }
    {
        size_t tot = (size_t)(1024/32)*(1024/16)*512;
        pack_b<<<dim3((unsigned)((tot+255)/256)), 256, 0, stream>>>(Wfc, 1024, nullptr, Wfc_pk, 1024, 1024, tot);
    }
    {
        size_t tot = (size_t)(1024/32)*(AP_/16)*512;
        pack_b<<<dim3((unsigned)((tot+255)/256)), 256, 0, stream>>>(Waf, 1024, nullptr, Waf_pk, 100, AP_, tot);
    }

    // init = tanh(mean @ W_feat_fc + b)  (64x1024)
    wmma_gemm<<<dim3(B_/16, (64+3)/4), 128, 0, stream>>>(
        fm_bf, C_, Wfc_pk, 64, bfc, nullptr, 0, init_f, nullptr, 1024, 1024, 1);
    init_split<<<B_, 512, 0, stream>>>(init_f, hf, cf, abuf);

    // feat_att = feats @ W_att_f + b  (16384 x 112)
    wmma_gemm<<<dim3((B_*HW_)/16, (AP_/16+3)/4), 128, 0, stream>>>(
        fb, C_, Waf_pk, AP_/16, bafp, nullptr, 0, fatt, nullptr, AP_, C_, 0);

    // ---------------- time loop ----------------
    for (int t = 0; t < T_; ++t) {
        step_attn<<<B_, 512, 0, stream>>>(hf, Wah, bah, Wav, bav, Wfb, bfb,
                                          fatt, ft, xb + (size_t)t*B_*E_, abuf,
                                          out_scores, t);
        // gates = [x|gated|h] @ [W_ih;W_hh] + (b_ih+b_hh)
        wmma_gemm<<<dim3(B_/16, 128/4), 128, 0, stream>>>(
            abuf, 2048, Wg_pk, 128, bias_g, nullptr, 0, gates, nullptr, 2048, 2048, 0);
        lstm_pw<<<B_, 512, 0, stream>>>(gates, hf, cf, abuf);
        // pre = tanh(x_t + [gated|h] @ [W_Lz;W_Lh] + b)  -> bf16
        wmma_gemm<<<dim3(B_/16, 32/4), 128, 0, stream>>>(
            abuf + 512, 2048, WL_pk, 32, bias_L, xf + (size_t)t*B_*E_, E_,
            nullptr, pre_bf, E_, 1536, 2);
        // logits = pre @ W_Lo + b_Lo  -> d_out[b, t, :]
        wmma_gemm<<<dim3(B_/16, (625+3)/4), 128, 0, stream>>>(
            pre_bf, E_, WLo_pk, 625, bLo, nullptr, 0,
            out_seq + (size_t)t * V_, nullptr, (long long)T_ * V_, E_, 0);
    }
}
